// MultiHeadSelfAttention_62027917689096
// MI455X (gfx1250) — matrix operand order repaired and checked by reading, not compiled, not run on hardware
//
#include <hip/hip_runtime.h>

#define D_EMB 384
#define SEQ   1024
#define NH    8
#define NB    8
#define HD    (NH * D_EMB)   // 3072

typedef __attribute__((ext_vector_type(16))) __bf16       v16bf;
typedef __attribute__((ext_vector_type(8)))  float        v8f;
typedef __attribute__((ext_vector_type(4)))  unsigned int v4u;

static __device__ __forceinline__ unsigned short f2bf(float f) {
  union { float f; unsigned u; } v; v.f = f;
  unsigned u = v.u;
  u += 0x7FFFu + ((u >> 16) & 1u);   // round-to-nearest-even
  return (unsigned short)(u >> 16);
}

// ---- WMMA fragment loaders (CDNA5 wave32 16x16x32 bf16 layouts) -------------
// A operand (16xK chunk of 32, row-major, leading dim ld):
// lane l<16: row l, K {0..7, 16..23}; lane l+16: row l, K {8..15, 24..31}.
static __device__ __forceinline__ v16bf load_a_bf16(const unsigned short* base, int ld, int lane) {
  int row = lane & 15;
  int kh  = (lane >> 4) * 8;
  union { v16bf v; v4u u[2]; } f;
  f.u[0] = *(const v4u*)(base + row * ld + kh);
  f.u[1] = *(const v4u*)(base + row * ld + kh + 16);
  return f.v;
}

// Same as above but source is f32; convert to bf16 in-register.
static __device__ __forceinline__ v16bf load_a_f32(const float* base, int ld, int lane) {
  int row = lane & 15;
  int kh  = (lane >> 4) * 8;
  const float4* p = (const float4*)(base + row * ld + kh);
  float4 a0 = p[0], a1 = p[1];
  const float4* q = (const float4*)(base + row * ld + kh + 16);
  float4 b0 = q[0], b1 = q[1];
  union { v16bf v; unsigned short s[16]; } f;
  f.s[0]=f2bf(a0.x); f.s[1]=f2bf(a0.y); f.s[2]=f2bf(a0.z); f.s[3]=f2bf(a0.w);
  f.s[4]=f2bf(a1.x); f.s[5]=f2bf(a1.y); f.s[6]=f2bf(a1.z); f.s[7]=f2bf(a1.w);
  f.s[8]=f2bf(b0.x); f.s[9]=f2bf(b0.y); f.s[10]=f2bf(b0.z); f.s[11]=f2bf(b0.w);
  f.s[12]=f2bf(b1.x); f.s[13]=f2bf(b1.y); f.s[14]=f2bf(b1.z); f.s[15]=f2bf(b1.w);
  return f.v;
}

// B operand (KxN = 32x16) read from "N-major" storage: base points at a block
// where row n (output column) is contiguous in K with leading dim ld.
// lane n<16: K {0..7, 16..23} of column n; lane n+16: K {8..15, 24..31}.
static __device__ __forceinline__ v16bf load_b_bf16(const unsigned short* base, int ld, int lane) {
  int col = lane & 15;
  int kr  = (lane >> 4) * 8;
  union { v16bf v; v4u u[2]; } f;
  f.u[0] = *(const v4u*)(base + col * ld + kr);
  f.u[1] = *(const v4u*)(base + col * ld + kr + 16);
  return f.v;
}

static __device__ __forceinline__ v8f wmma_bf16(v16bf a, v16bf b, v8f c) {
  return __builtin_amdgcn_wmma_f32_16x16x32_bf16(false, a, false, b, (short)0, c, false, false);
}

// ---- Kernel 0: transpose + bf16-convert weights -----------------------------
// Wq/Wk/Wv: [H][Din][Dout] f32 -> W*t: [H][Dout][Din] bf16
// Wo: [HD][D] f32 -> Wot: [D][HD] bf16
__global__ __launch_bounds__(256) void prep_weights_kernel(
    const float* __restrict__ Wq, const float* __restrict__ Wk,
    const float* __restrict__ Wv, const float* __restrict__ Wo,
    unsigned short* __restrict__ Wqt, unsigned short* __restrict__ Wkt,
    unsigned short* __restrict__ Wvt, unsigned short* __restrict__ Wot) {
  int i = blockIdx.x * blockDim.x + threadIdx.x;
  const int HW = NH * D_EMB * D_EMB;
  if (i < 3 * HW) {
    int which = i / HW, rem = i % HW;
    int h = rem / (D_EMB * D_EMB), t = rem % (D_EMB * D_EMB);
    int dout = t / D_EMB, din = t % D_EMB;
    const float* src = (which == 0) ? Wq : (which == 1) ? Wk : Wv;
    unsigned short* dst = (which == 0) ? Wqt : (which == 1) ? Wkt : Wvt;
    dst[h * D_EMB * D_EMB + dout * D_EMB + din] =
        f2bf(src[h * D_EMB * D_EMB + din * D_EMB + dout]);
  } else {
    int j = i - 3 * HW;
    if (j < D_EMB * HD) {
      int dout = j / HD, din = j % HD;
      Wot[dout * HD + din] = f2bf(Wo[din * D_EMB + dout]);
    }
  }
}

// ---- Kernel 1: QKV projections ----------------------------------------------
// One wave computes a 16(s) x 64(d) strip of one of {Q,K,V}[h][b].
// Q,K stored row-major [S][D] bf16; V stored transposed [D][S] bf16.
__global__ __launch_bounds__(128) void qkv_proj_kernel(
    const float* __restrict__ x,
    const float* __restrict__ bq, const float* __restrict__ bk, const float* __restrict__ bv,
    const unsigned short* __restrict__ Wqt, const unsigned short* __restrict__ Wkt,
    const unsigned short* __restrict__ Wvt,
    unsigned short* __restrict__ Q, unsigned short* __restrict__ K,
    unsigned short* __restrict__ Vt) {
  int lane = threadIdx.x & 31;
  int strip = blockIdx.x * 4 + (threadIdx.x >> 5);   // 3*8*8*64*6 = 73728 strips
  int nblk = strip % 6;  strip /= 6;
  int sblk = strip % 64; strip /= 64;
  int b    = strip % 8;  strip /= 8;
  int h    = strip % 8;  strip /= 8;
  int which = strip;                                  // 0=Q 1=K 2=V
  const unsigned short* Wt = (which == 0) ? Wqt : (which == 1) ? Wkt : Wvt;
  const float* bias = (which == 0) ? bq : (which == 1) ? bk : bv;
  Wt   += h * D_EMB * D_EMB;
  bias += h * D_EMB;
  int s0 = sblk * 16, n0 = nblk * 64;
  const float* xrow = x + ((size_t)b * SEQ + s0) * D_EMB;

  v8f acc[4] = {};
  #pragma unroll
  for (int k0 = 0; k0 < D_EMB; k0 += 32) {
    v16bf a = load_a_f32(xrow + k0, D_EMB, lane);
    v16bf bb[4];
    #pragma unroll
    for (int j = 0; j < 4; j++)
      bb[j] = load_b_bf16(Wt + (n0 + j * 16) * D_EMB + k0, D_EMB, lane);
    #pragma unroll
    for (int j = 0; j < 4; j++)
      acc[j] = wmma_bf16(a, bb[j], acc[j]);
  }

  int col = lane & 15, half = lane >> 4;
  size_t slice = (size_t)(h * NB + b) * SEQ * D_EMB;
  #pragma unroll
  for (int j = 0; j < 4; j++) {
    int d = n0 + j * 16 + col;
    float bv_ = bias[d];
    if (which < 2) {
      unsigned short* out = ((which == 0) ? Q : K) + slice;
      #pragma unroll
      for (int r = 0; r < 8; r++) {
        int s = s0 + half * 8 + r;                       // C layout: M = r + 8*half
        out[(size_t)s * D_EMB + d] = f2bf(acc[j][r] + bv_);
      }
    } else {
      unsigned short* out = Vt + slice;                  // [D][S]
      union { v4u u; unsigned short s[8]; } pk;
      #pragma unroll
      for (int r = 0; r < 8; r++) pk.s[r] = f2bf(acc[j][r] + bv_);
      *(v4u*)(out + (size_t)d * SEQ + s0 + half * 8) = pk.u;  // 8 consecutive s
    }
  }
}

// ---- Kernel 2: attention (scores + softmax + P*V) ---------------------------
// One 256-thread WG per (h, b, 16-row q block). Dynamic LDS: 96 KB
// (64 KB f32 score panel [16][1024] + 32 KB bf16 P panel [16][1024]).
__global__ __launch_bounds__(256) void attention_kernel(
    const unsigned short* __restrict__ Q, const unsigned short* __restrict__ K,
    const unsigned short* __restrict__ Vt, const int* __restrict__ mask,
    unsigned short* __restrict__ concat) {
  extern __shared__ char smem[];
  float* Ssc = (float*)smem;                                   // [16][SEQ]
  unsigned short* P = (unsigned short*)(smem + 16 * SEQ * 4);  // [16][SEQ]

  int lane = threadIdx.x & 31, wid = threadIdx.x >> 5;
  int qblk = blockIdx.x & 63;
  int b    = (blockIdx.x >> 6) & 7;
  int h    = blockIdx.x >> 9;
  int s0   = qblk * 16;
  size_t slice = (size_t)(h * NB + b) * SEQ * D_EMB;
  const unsigned short* Qb = Q + slice;
  const unsigned short* Kb = K + slice;
  const unsigned short* Vb = Vt + slice;
  int nkt = qblk + 1;                      // key tiles of 16 (causal)
  const float scale = 0.051031036307982884f;  // 1/sqrt(384)
  int col = lane & 15, half = lane >> 4;

  // Hoist the Q row-panel fragments (reused for every key tile).
  v16bf qa[12];
  #pragma unroll
  for (int kk = 0; kk < 12; kk++)
    qa[kk] = load_a_bf16(Qb + (size_t)s0 * D_EMB + kk * 32, D_EMB, lane);

  // Phase 1: score tiles, scaled + causal/pad masked, into LDS.
  for (int kt = wid; kt < nkt; kt += 8) {
    const unsigned short* Krow = Kb + (size_t)(kt * 16) * D_EMB;
    v8f c = {};
    #pragma unroll
    for (int kc = 0; kc < 3; kc++) {           // 3 groups of 4 batched B loads
      v16bf bb[4];
      #pragma unroll
      for (int j = 0; j < 4; j++)
        bb[j] = load_b_bf16(Krow + (kc * 4 + j) * 32, D_EMB, lane);
      #pragma unroll
      for (int j = 0; j < 4; j++)
        c = wmma_bf16(qa[kc * 4 + j], bb[j], c);
    }
    int key = kt * 16 + col;
    int mk  = mask[b * SEQ + key];
    #pragma unroll
    for (int r = 0; r < 8; r++) {
      int m = half * 8 + r;                 // local q row
      bool valid = (key <= s0 + m) && (mk != 0);
      Ssc[m * SEQ + key] = valid ? c[r] * scale : -10000.0f;
    }
  }
  __syncthreads();

  // Phase 2: softmax, 16 threads per row; write bf16 P (disjoint LDS region).
  {
    int r = threadIdx.x >> 4;
    int i = threadIdx.x & 15;
    int kmax = nkt * 16;
    const float* row = Ssc + r * SEQ;
    float m = -3.0e38f;
    for (int k = i; k < kmax; k += 16) m = fmaxf(m, row[k]);
    #pragma unroll
    for (int off = 8; off >= 1; off >>= 1) m = fmaxf(m, __shfl_xor(m, off));
    float sum = 0.0f;
    for (int k = i; k < kmax; k += 16) sum += __expf(row[k] - m);
    #pragma unroll
    for (int off = 8; off >= 1; off >>= 1) sum += __shfl_xor(sum, off);
    float inv = 1.0f / sum;
    unsigned short* prow = P + r * SEQ;
    for (int k = i; k < kmax; k += 16) prow[k] = f2bf(__expf(row[k] - m) * inv);
    if (nkt & 1) prow[kmax + i] = 0;        // zero-pad to a 32-key multiple
  }
  __syncthreads();

  // Phase 3: O = P * V over 32-key chunks, 2-wide pipelined; concat bf16 out.
  int nk32 = (nkt + 1) >> 1;
  for (int dt = wid; dt < 24; dt += 8) {
    const unsigned short* Vrow = Vb + (size_t)(dt * 16) * SEQ;
    v8f c = {};
    int kc = 0;
    for (; kc + 2 <= nk32; kc += 2) {
      v16bf a0 = load_a_bf16(P + kc * 32, SEQ, lane);
      v16bf a1 = load_a_bf16(P + (kc + 1) * 32, SEQ, lane);
      v16bf b0 = load_b_bf16(Vrow + kc * 32, SEQ, lane);
      v16bf b1 = load_b_bf16(Vrow + (kc + 1) * 32, SEQ, lane);
      c = wmma_bf16(a0, b0, c);
      c = wmma_bf16(a1, b1, c);
    }
    if (kc < nk32) {
      v16bf a0 = load_a_bf16(P + kc * 32, SEQ, lane);
      v16bf b0 = load_b_bf16(Vrow + kc * 32, SEQ, lane);
      c = wmma_bf16(a0, b0, c);
    }
    int d = dt * 16 + col;
    #pragma unroll
    for (int r = 0; r < 8; r++) {
      int s = s0 + half * 8 + r;
      concat[(size_t)(b * SEQ + s) * HD + h * D_EMB + d] = f2bf(c[r]);
    }
  }
}

// ---- Kernel 3: output projection + bias + mask ------------------------------
// One wave computes a 16(m) x 64(n) strip of out = concat @ Wo + bo, * mask.
__global__ __launch_bounds__(128) void out_proj_kernel(
    const unsigned short* __restrict__ concat, const unsigned short* __restrict__ Wot,
    const float* __restrict__ bo, const int* __restrict__ mask,
    float* __restrict__ out) {
  int lane = threadIdx.x & 31;
  int strip = blockIdx.x * 4 + (threadIdx.x >> 5);   // 512*6 = 3072 strips
  int nblk = strip % 6;
  int mblk = strip / 6;
  int m0 = mblk * 16, n0 = nblk * 64;
  int b  = m0 / SEQ, s0 = m0 % SEQ;
  const unsigned short* arow = concat + (size_t)m0 * HD;

  v8f acc[4] = {};
  #pragma unroll 2
  for (int k0 = 0; k0 < HD; k0 += 32) {
    v16bf a = load_a_bf16(arow + k0, HD, lane);
    v16bf bb[4];
    #pragma unroll
    for (int j = 0; j < 4; j++)
      bb[j] = load_b_bf16(Wot + (size_t)(n0 + j * 16) * HD + k0, HD, lane);
    #pragma unroll
    for (int j = 0; j < 4; j++)
      acc[j] = wmma_bf16(a, bb[j], acc[j]);
  }

  int col = lane & 15, half = lane >> 4;
  #pragma unroll
  for (int j = 0; j < 4; j++) {
    int d = n0 + j * 16 + col;
    float bias = bo[d];
    #pragma unroll
    for (int r = 0; r < 8; r++) {
      int s = s0 + half * 8 + r;
      float mv = (float)mask[b * SEQ + s];
      out[(size_t)(b * SEQ + s) * D_EMB + d] = (acc[j][r] + bias) * mv;
    }
  }
}

// ---- Host launch ------------------------------------------------------------
extern "C" void kernel_launch(void* const* d_in, const int* in_sizes, int n_in,
                              void* d_out, int out_size, void* d_ws, size_t ws_size,
                              hipStream_t stream) {
  (void)in_sizes; (void)n_in; (void)out_size; (void)ws_size;
  const float* x  = (const float*)d_in[0];
  const int* mask = (const int*)d_in[1];
  const float* Wq = (const float*)d_in[2];
  const float* bq = (const float*)d_in[3];
  const float* Wk = (const float*)d_in[4];
  const float* bk = (const float*)d_in[5];
  const float* Wv = (const float*)d_in[6];
  const float* bv = (const float*)d_in[7];
  const float* Wo = (const float*)d_in[8];
  const float* bo = (const float*)d_in[9];
  float* out = (float*)d_out;

  const size_t WT_B   = (size_t)NH * D_EMB * D_EMB * 2;    // 2,359,296
  const size_t WOT_B  = (size_t)D_EMB * HD * 2;            // 2,359,296
  const size_t QKV_B  = (size_t)NH * NB * SEQ * D_EMB * 2; // 50,331,648
  const size_t CCAT_B = (size_t)NB * SEQ * HD * 2;         // 50,331,648
  char* ws = (char*)d_ws;
  size_t off = 0;
  unsigned short* Wqt   = (unsigned short*)(ws + off); off += WT_B;
  unsigned short* Wkt   = (unsigned short*)(ws + off); off += WT_B;
  unsigned short* Wvt   = (unsigned short*)(ws + off); off += WT_B;
  unsigned short* Wot   = (unsigned short*)(ws + off); off += WOT_B;
  unsigned short* Qws   = (unsigned short*)(ws + off); off += QKV_B;
  unsigned short* Kws   = (unsigned short*)(ws + off); off += QKV_B;
  unsigned short* Vtws  = (unsigned short*)(ws + off); off += QKV_B;
  unsigned short* Ccat  = (unsigned short*)(ws + off); off += CCAT_B;

  (void)hipFuncSetAttribute((const void*)attention_kernel,
                            hipFuncAttributeMaxDynamicSharedMemorySize, 96 * 1024);

  // 0) transpose+convert weights: 3*8*384*384 + 384*3072 = 4,718,592 elements
  prep_weights_kernel<<<18432, 256, 0, stream>>>(Wq, Wk, Wv, Wo, Wqt, Wkt, Wvt, Wot);
  // 1) QKV projections: 9216 wave-strips, 4 waves/block
  qkv_proj_kernel<<<2304, 128, 0, stream>>>(x, bq, bk, bv, Wqt, Wkt, Wvt, Qws, Kws, Vtws);
  // 2) attention: one WG per (h,b,qblock16) = 8*8*64
  attention_kernel<<<4096, 256, 96 * 1024, stream>>>(Qws, Kws, Vtws, mask, Ccat);
  // 3) output projection: 3072 wave-strips, 4 waves/block
  out_proj_kernel<<<768, 128, 0, stream>>>(Ccat, Wot, bo, mask, out);
}
